// TransformerBlock_69939247448356
// MI455X (gfx1250) — compile-verified, matrix operand order checked by reading, not run on hardware
//
#include <hip/hip_runtime.h>
#include <hip/hip_bf16.h>

typedef _Float16 half_t;
typedef __attribute__((ext_vector_type(16))) _Float16 v16h;
typedef __attribute__((ext_vector_type(8)))  _Float16 v8h;
typedef __attribute__((ext_vector_type(8)))  float    v8f;

// ---------------------------------------------------------------------------
// WMMA fragment helpers (CDNA5 16x16x32 f16, wave32)
// A-fragment per ISA 7.12.2: lane (m = lane&15, c0 = (lane&16)?8:0) holds
// elements 0..7  = row[m][c0 .. c0+7]
// elements 8..15 = row[m][c0+16 .. c0+23]
// B-fragment (K-major operand, lane n = output column) loads identically.
// ---------------------------------------------------------------------------
__device__ __forceinline__ v16h frag_ld(const half_t* rowp, int c0) {
    v8h lo = *(const v8h*)(rowp + c0);
    v8h hi = *(const v8h*)(rowp + c0 + 16);
    return __builtin_shufflevector(lo, hi, 0,1,2,3,4,5,6,7,8,9,10,11,12,13,14,15);
}

__device__ __forceinline__ v8f wmma_f16(v16h a, v16h b, v8f c) {
    return __builtin_amdgcn_wmma_f32_16x16x32_f16(false, a, false, b, (short)0, c, false, false);
}

__device__ __forceinline__ v8f v8f_zero() {
    v8f z = {0.f,0.f,0.f,0.f,0.f,0.f,0.f,0.f};
    return z;
}

__device__ __forceinline__ float gelu_exact(float v) {
    return 0.5f * v * (1.0f + erff(v * 0.70710678118654752f));
}

// ---------------------------------------------------------------------------
// CDNA5 async global->LDS copy (ASYNCcnt-tracked, no VGPR staging).
// LDS operand is the low 32 bits of the generic pointer (ISA 10.2: LDS
// aperture keeps the byte offset in addr[31:0]).
// ---------------------------------------------------------------------------
__device__ __forceinline__ void async_copy_b128(void* lds_dst, const void* gsrc) {
    unsigned l = (unsigned)(unsigned long long)lds_dst;
    asm volatile("global_load_async_to_lds_b128 %0, %1, off"
                 :: "v"(l), "v"(gsrc) : "memory");
}
__device__ __forceinline__ void wait_async_le4() {
    asm volatile("s_wait_asynccnt 0x4" ::: "memory");
}
__device__ __forceinline__ void wait_async_0() {
    asm volatile("s_wait_asynccnt 0x0" ::: "memory");
}

// ---------------------------------------------------------------------------
// Elementwise f32 -> f16 (weight conversion)
// ---------------------------------------------------------------------------
__global__ __launch_bounds__(256) void f32_to_f16_kernel(
    const float* __restrict__ in, half_t* __restrict__ out, int n) {
    int i = blockIdx.x * 256 + threadIdx.x;
    if (i < n) out[i] = (half_t)in[i];
}

// ---------------------------------------------------------------------------
// LayerNorm over last dim (768), one token per block (256 threads), f16 out
// ---------------------------------------------------------------------------
__global__ __launch_bounds__(256) void ln_kernel(
    const float* __restrict__ x, const float* __restrict__ w,
    const float* __restrict__ b, half_t* __restrict__ out) {
    __shared__ float red[256];
    __shared__ float stat[2];
    const int t = threadIdx.x;
    const size_t tok = blockIdx.x;
    const float* xp = x + tok * 768;
    float v0 = xp[t], v1 = xp[t + 256], v2 = xp[t + 512];
    red[t] = v0 + v1 + v2;
    __syncthreads();
    #pragma unroll
    for (int k = 128; k > 0; k >>= 1) { if (t < k) red[t] += red[t + k]; __syncthreads(); }
    if (t == 0) stat[0] = red[0] * (1.0f / 768.0f);
    __syncthreads();
    const float mu = stat[0];
    float d0 = v0 - mu, d1 = v1 - mu, d2 = v2 - mu;
    red[t] = d0 * d0 + d1 * d1 + d2 * d2;
    __syncthreads();
    #pragma unroll
    for (int k = 128; k > 0; k >>= 1) { if (t < k) red[t] += red[t + k]; __syncthreads(); }
    if (t == 0) stat[1] = rsqrtf(red[0] * (1.0f / 768.0f) + 1e-5f);
    __syncthreads();
    const float rs = stat[1];
    half_t* op = out + tok * 768;
    op[t]       = (half_t)(d0 * rs * w[t]       + b[t]);
    op[t + 256] = (half_t)(d1 * rs * w[t + 256] + b[t + 256]);
    op[t + 512] = (half_t)(d2 * rs * w[t + 512] + b[t + 512]);
}

// ---------------------------------------------------------------------------
// WMMA GEMM:  C[M,N] = A[M,K] * W[N,K]^T   (both K-major f16), f32 accumulate
// Block tile 128x128, K-tile 32, double-buffered LDS filled by
// global_load_async_to_lds_b128 (ASYNCcnt pipeline). 256 threads = 8 waves;
// wave tile 64x32 (4 M-tiles x 2 N-tiles of 16x16).
// EPI: 1 = QKV head scatter (f16), 2 = bias+residual (f32 out),
//      3 = bias+GELU (f16 out), 4 = bias+residual (f32 out, final)
// ---------------------------------------------------------------------------
template<int EPI>
__global__ __launch_bounds__(256, 2) void gemm_kernel(
    const half_t* __restrict__ A, const half_t* __restrict__ W,
    const float* __restrict__ bias, const float* __restrict__ resid,
    float* __restrict__ outF, half_t* __restrict__ outH,
    int M, int N, int K) {
    constexpr int LDT = 48;                 // LDS row stride in halves (96B, 16B-aligned)
    __shared__ half_t As[2][128 * LDT];
    __shared__ half_t Bs[2][128 * LDT];

    const int bm = blockIdx.y, bn = blockIdx.x;
    const int tid = threadIdx.x;
    const int wave = tid >> 5, lane = tid & 31;
    const int wm = wave & 1;                // 0..1 -> 64-row band
    const int wn = wave >> 1;               // 0..3 -> 32-col band
    const int lm = lane & 15;
    const int c0 = (lane & 16) ? 8 : 0;
    const int hfid = lane >> 4;
    const int Kq = K >> 3;                  // K in uint4 units

    v8f acc[4][2];
    #pragma unroll
    for (int mt = 0; mt < 4; ++mt)
        #pragma unroll
        for (int nt = 0; nt < 2; ++nt) acc[mt][nt] = v8f_zero();

    const uint4* Ag = (const uint4*)(A + (size_t)bm * 128 * K);
    const uint4* Wg = (const uint4*)(W + (size_t)bn * 128 * K);

    // issue one 128x32 A tile + 128x32 B tile as 4 async b128 copies/thread
    auto issue_tile = [&](int k0, int buf) {
        const int kq0 = k0 >> 3;
        #pragma unroll
        for (int i = 0; i < 2; ++i) {       // 512 uint4 per matrix / 256 threads
            int chunk = tid + i * 256;
            int r = chunk >> 2, c = chunk & 3;
            async_copy_b128(&As[buf][r * LDT + c * 8], &Ag[(size_t)r * Kq + kq0 + c]);
            async_copy_b128(&Bs[buf][r * LDT + c * 8], &Wg[(size_t)r * Kq + kq0 + c]);
        }
    };

    const int nk = K >> 5;                  // number of 32-wide K tiles
    issue_tile(0, 0);
    for (int t = 0; t < nk; ++t) {
        const int buf = t & 1;
        if (t + 1 < nk) {
            issue_tile((t + 1) << 5, buf ^ 1);
            wait_async_le4();               // tile t's 4 copies complete (in-order)
        } else {
            wait_async_0();
        }
        __syncthreads();

        v16h af[4], bf[2];
        #pragma unroll
        for (int mt = 0; mt < 4; ++mt)
            af[mt] = frag_ld(&As[buf][(wm * 64 + mt * 16 + lm) * LDT], c0);
        #pragma unroll
        for (int nt = 0; nt < 2; ++nt)
            bf[nt] = frag_ld(&Bs[buf][(wn * 32 + nt * 16 + lm) * LDT], c0);
        #pragma unroll
        for (int mt = 0; mt < 4; ++mt)
            #pragma unroll
            for (int nt = 0; nt < 2; ++nt)
                acc[mt][nt] = wmma_f16(af[mt], bf[nt], acc[mt][nt]);
        __syncthreads();
    }

    // Epilogue. C/D layout: lane (n = lane&15, half = lane>>4), element r is
    // C[row = r + 8*half][col = n] of the 16x16 tile.
    #pragma unroll
    for (int mt = 0; mt < 4; ++mt) {
        #pragma unroll
        for (int nt = 0; nt < 2; ++nt) {
            const int col = bn * 128 + wn * 32 + nt * 16 + lm;
            #pragma unroll
            for (int r = 0; r < 8; ++r) {
                const int row = bm * 128 + wm * 64 + mt * 16 + r + 8 * hfid;
                float v = acc[mt][nt][r];
                if constexpr (EPI == 1) {
                    // scatter into qkv[which][b][h][s][d], B=4,H=12,S=2048,D=64
                    int which = col / 768, rem = col % 768;
                    int hh = rem >> 6, dd = rem & 63;
                    int bb = row >> 11, ss = row & 2047;
                    outH[((((size_t)which * 4 + bb) * 12 + hh) * 2048 + ss) * 64 + dd] = (half_t)v;
                } else if constexpr (EPI == 2) {
                    v += bias[col] + resid[(size_t)row * N + col];
                    outF[(size_t)row * N + col] = v;
                } else if constexpr (EPI == 3) {
                    v = gelu_exact(v + bias[col]);
                    outH[(size_t)row * N + col] = (half_t)v;
                } else { // EPI == 4
                    v += bias[col] + resid[(size_t)row * N + col];
                    outF[(size_t)row * N + col] = v;
                }
            }
        }
    }
}

// ---------------------------------------------------------------------------
// Flash attention: one block = 128 threads = 4 waves; each wave owns 32 q-rows.
// Iterates over keys in chunks of 64; QK^T and P*V via WMMA; online softmax.
// K chunk staged with async global->LDS copies; V chunk transposed via
// coalesced b128 global reads + b16 LDS scatter.
// qkv layout: [which(3)][b(4)][h(12)][s(2048)][d(64)] f16.
// Output: [b][s][h*64+d] f16 (E = 768).
// ---------------------------------------------------------------------------
__global__ __launch_bounds__(128) void attn_kernel(
    const half_t* __restrict__ qkv, half_t* __restrict__ outh) {
    constexpr int S = 2048, D = 64, LK = 72;    // 144B LDS stride (16B-aligned)
    __shared__ half_t Ks[64 * LK];
    __shared__ half_t Vt[64 * LK];              // transposed V chunk: [d][key]
    __shared__ half_t Ps[4][32 * LK];           // per-wave P staging

    const int tid = threadIdx.x;
    const int wave = tid >> 5, lane = tid & 31;
    const int lm = lane & 15, hf = lane >> 4;
    const int c0 = (lane & 16) ? 8 : 0;
    const int bh = blockIdx.y;                  // b*12 + h, 0..47
    const int q0 = blockIdx.x * 128 + wave * 32;

    const half_t* Qp = qkv + (size_t)bh * S * D;
    const half_t* Kp = qkv + (size_t)(48 + bh) * S * D;
    const half_t* Vp = qkv + (size_t)(96 + bh) * S * D;

    // Q fragments held in registers: 2 m-tiles x 2 k-steps over D=64
    v16h qf[2][2];
    #pragma unroll
    for (int mt = 0; mt < 2; ++mt)
        #pragma unroll
        for (int ks = 0; ks < 2; ++ks)
            qf[mt][ks] = frag_ld(Qp + (size_t)(q0 + mt * 16 + lm) * D + ks * 32, c0);

    float mstat[2][8], lstat[2][8];
    v8f o[2][4];
    #pragma unroll
    for (int mt = 0; mt < 2; ++mt) {
        #pragma unroll
        for (int r = 0; r < 8; ++r) { mstat[mt][r] = -1e30f; lstat[mt][r] = 0.f; }
        #pragma unroll
        for (int dt = 0; dt < 4; ++dt) o[mt][dt] = v8f_zero();
    }

    for (int kc = 0; kc < S; kc += 64) {
        // async-stage K chunk (row-major): 512 uint4 / 128 threads
        const uint4* Kg = (const uint4*)(Kp + (size_t)kc * D);
        #pragma unroll
        for (int i = 0; i < 4; ++i) {
            int chunk = tid + i * 128;
            int r = chunk >> 3, c = chunk & 7;
            async_copy_b128(&Ks[r * LK + c * 8], &Kg[chunk]);
        }
        // V chunk: coalesced b128 reads, transpose into Vt via b16 LDS stores
        const uint4* Vg = (const uint4*)(Vp + (size_t)kc * D);
        #pragma unroll
        for (int i = 0; i < 4; ++i) {
            int chunk = tid + i * 128;
            int key = chunk >> 3, c = chunk & 7;
            uint4 vv = Vg[chunk];
            half_t tmp[8];
            *(uint4*)tmp = vv;
            #pragma unroll
            for (int j = 0; j < 8; ++j)
                Vt[(c * 8 + j) * LK + key] = tmp[j];
        }
        wait_async_0();
        __syncthreads();

        // S = Q K^T (scaled)
        v8f sc[2][4];
        #pragma unroll
        for (int mt = 0; mt < 2; ++mt)
            #pragma unroll
            for (int nt = 0; nt < 4; ++nt) sc[mt][nt] = v8f_zero();
        #pragma unroll
        for (int ks = 0; ks < 2; ++ks) {
            v16h kf[4];
            #pragma unroll
            for (int nt = 0; nt < 4; ++nt)
                kf[nt] = frag_ld(&Ks[(nt * 16 + lm) * LK + ks * 32], c0);
            #pragma unroll
            for (int mt = 0; mt < 2; ++mt)
                #pragma unroll
                for (int nt = 0; nt < 4; ++nt)
                    sc[mt][nt] = wmma_f16(qf[mt][ks], kf[nt], sc[mt][nt]);
        }

        // online softmax per 8 rows owned by this lane
        #pragma unroll
        for (int mt = 0; mt < 2; ++mt) {
            float rm[8];
            #pragma unroll
            for (int r = 0; r < 8; ++r) {
                float mx = -1e30f;
                #pragma unroll
                for (int nt = 0; nt < 4; ++nt) {
                    sc[mt][nt][r] *= 0.125f;    // HEAD_DIM^-0.5
                    mx = fmaxf(mx, sc[mt][nt][r]);
                }
                rm[r] = mx;
            }
            #pragma unroll
            for (int mask = 8; mask >= 1; mask >>= 1)
                #pragma unroll
                for (int r = 0; r < 8; ++r)
                    rm[r] = fmaxf(rm[r], __shfl_xor(rm[r], mask, 32));
            float alpha[8], rs[8];
            #pragma unroll
            for (int r = 0; r < 8; ++r) {
                float mn = fmaxf(mstat[mt][r], rm[r]);
                alpha[r] = __expf(mstat[mt][r] - mn);
                mstat[mt][r] = mn;
                rs[r] = 0.f;
            }
            #pragma unroll
            for (int nt = 0; nt < 4; ++nt)
                #pragma unroll
                for (int r = 0; r < 8; ++r) {
                    float p = __expf(sc[mt][nt][r] - mstat[mt][r]);
                    sc[mt][nt][r] = p;
                    rs[r] += p;
                }
            #pragma unroll
            for (int mask = 8; mask >= 1; mask >>= 1)
                #pragma unroll
                for (int r = 0; r < 8; ++r)
                    rs[r] += __shfl_xor(rs[r], mask, 32);
            #pragma unroll
            for (int r = 0; r < 8; ++r)
                lstat[mt][r] = lstat[mt][r] * alpha[r] + rs[r];
            #pragma unroll
            for (int dt = 0; dt < 4; ++dt)
                #pragma unroll
                for (int r = 0; r < 8; ++r)
                    o[mt][dt][r] *= alpha[r];
            // stage P (f16) into per-wave LDS region for A-fragment reload
            #pragma unroll
            for (int nt = 0; nt < 4; ++nt)
                #pragma unroll
                for (int r = 0; r < 8; ++r)
                    Ps[wave][(mt * 16 + r + 8 * hf) * LK + nt * 16 + lm] =
                        (half_t)sc[mt][nt][r];
        }

        // O += P * V
        #pragma unroll
        for (int ks = 0; ks < 2; ++ks) {
            v16h pf[2], vf[4];
            #pragma unroll
            for (int mt = 0; mt < 2; ++mt)
                pf[mt] = frag_ld(&Ps[wave][(mt * 16 + lm) * LK + ks * 32], c0);
            #pragma unroll
            for (int dt = 0; dt < 4; ++dt)
                vf[dt] = frag_ld(&Vt[(dt * 16 + lm) * LK + ks * 32], c0);
            #pragma unroll
            for (int mt = 0; mt < 2; ++mt)
                #pragma unroll
                for (int dt = 0; dt < 4; ++dt)
                    o[mt][dt] = wmma_f16(pf[mt], vf[dt], o[mt][dt]);
        }
        __syncthreads();
    }

    // finalize: divide by softmax denominator and scatter into (B,S,E) f16
    const int bb = bh / 12, hh = bh % 12;
    #pragma unroll
    for (int mt = 0; mt < 2; ++mt) {
        float inv[8];
        #pragma unroll
        for (int r = 0; r < 8; ++r) inv[r] = 1.0f / lstat[mt][r];
        #pragma unroll
        for (int dt = 0; dt < 4; ++dt)
            #pragma unroll
            for (int r = 0; r < 8; ++r) {
                int row = q0 + mt * 16 + r + 8 * hf;
                int col = hh * 64 + dt * 16 + lm;
                outh[((size_t)bb * S + row) * 768 + col] = (half_t)(o[mt][dt][r] * inv[r]);
            }
    }
}

// ---------------------------------------------------------------------------
// Host-side orchestration
// ---------------------------------------------------------------------------
extern "C" void kernel_launch(void* const* d_in, const int* in_sizes, int n_in,
                              void* d_out, int out_size, void* d_ws, size_t ws_size,
                              hipStream_t stream) {
    (void)in_sizes; (void)n_in; (void)out_size; (void)ws_size;
    const float* x      = (const float*)d_in[0];
    const float* ln1_w  = (const float*)d_in[1];
    const float* ln1_b  = (const float*)d_in[2];
    const float* ln2_w  = (const float*)d_in[3];
    const float* ln2_b  = (const float*)d_in[4];
    const float* qkv_w  = (const float*)d_in[5];
    const float* proj_w = (const float*)d_in[6];
    const float* proj_b = (const float*)d_in[7];
    const float* fc1_w  = (const float*)d_in[8];
    const float* fc1_b  = (const float*)d_in[9];
    const float* fc2_w  = (const float*)d_in[10];
    const float* fc2_b  = (const float*)d_in[11];
    float* out = (float*)d_out;

    const int M = 4 * 2048;           // 8192 tokens
    const int E = 768, HID = 3072, QKV = 2304;

    char* ws = (char*)d_ws;
    size_t off = 0;
    auto alloc = [&](size_t bytes) -> char* {
        char* p = ws + off;
        off = (off + bytes + 255) & ~(size_t)255;
        return p;
    };
    half_t* qkv_wh  = (half_t*)alloc((size_t)QKV * E * 2);
    half_t* proj_wh = (half_t*)alloc((size_t)E * E * 2);
    half_t* fc1_wh  = (half_t*)alloc((size_t)HID * E * 2);
    half_t* fc2_wh  = (half_t*)alloc((size_t)E * HID * 2);
    half_t* ln1h    = (half_t*)alloc((size_t)M * E * 2);
    half_t* qkvh    = (half_t*)alloc((size_t)M * QKV * 2);
    half_t* attnh   = (half_t*)alloc((size_t)M * E * 2);
    float*  y1      = (float*)alloc((size_t)M * E * 4);
    half_t* ln2h    = (half_t*)alloc((size_t)M * E * 2);
    half_t* hh      = (half_t*)alloc((size_t)M * HID * 2);

    auto cvt = [&](const float* src, half_t* dst, int n) {
        f32_to_f16_kernel<<<(n + 255) / 256, 256, 0, stream>>>(src, dst, n);
    };
    cvt(qkv_w,  qkv_wh,  QKV * E);
    cvt(proj_w, proj_wh, E * E);
    cvt(fc1_w,  fc1_wh,  HID * E);
    cvt(fc2_w,  fc2_wh,  E * HID);

    // x -> LN1 -> f16
    ln_kernel<<<M, 256, 0, stream>>>(x, ln1_w, ln1_b, ln1h);
    // QKV projection with head scatter
    gemm_kernel<1><<<dim3(QKV / 128, M / 128), 256, 0, stream>>>(
        ln1h, qkv_wh, nullptr, nullptr, nullptr, qkvh, M, QKV, E);
    // flash attention
    attn_kernel<<<dim3(2048 / 128, 48), 128, 0, stream>>>(qkvh, attnh);
    // output projection + bias + residual(x) -> y1 (f32)
    gemm_kernel<2><<<dim3(E / 128, M / 128), 256, 0, stream>>>(
        attnh, proj_wh, proj_b, x, y1, nullptr, M, E, E);
    // y1 -> LN2 -> f16
    ln_kernel<<<M, 256, 0, stream>>>(y1, ln2_w, ln2_b, ln2h);
    // FC1 + bias + GELU -> f16
    gemm_kernel<3><<<dim3(HID / 128, M / 128), 256, 0, stream>>>(
        ln2h, fc1_wh, fc1_b, nullptr, nullptr, hh, M, HID, E);
    // FC2 + bias + residual(y1) -> final f32 output
    gemm_kernel<4><<<dim3(E / 128, M / 128), 256, 0, stream>>>(
        hh, fc2_wh, fc2_b, y1, out, nullptr, M, E, HID);
}
